// DualRadarFusion_42262478193380
// MI455X (gfx1250) — matrix operand order repaired and checked by reading, not compiled, not run on hardware
//
#include <hip/hip_runtime.h>
#include <hip/hip_bf16.h>
#include <math.h>

// ---------------- problem constants ----------------
constexpr int nB = 8, nCIN = 16, nL = 256, nD = 256, nH = 8;
constexpr int nDK = 32, nDV = 32, nK = 3, nKE = 5;
constexpr int nL4 = 4 * nL;      // 1024
constexpr int nHD = nH * nDK;    // 256

typedef __bf16 bf16;
typedef __attribute__((ext_vector_type(2)))  __bf16 v2bf;
typedef __attribute__((ext_vector_type(16))) __bf16 v16bf;
typedef __attribute__((ext_vector_type(8)))  float  v8f;

// TDM descriptor register groups (see CDNA5 ISA)
typedef __attribute__((ext_vector_type(4))) unsigned int tdm_u4;
typedef __attribute__((ext_vector_type(8))) int          tdm_i8;
typedef __attribute__((ext_vector_type(4))) int          tdm_i4;

#if __has_builtin(__builtin_amdgcn_tensor_load_to_lds) && \
    __has_builtin(__builtin_amdgcn_s_wait_tensorcnt)
#define USE_TDM_STAGE 1
#else
#define USE_TDM_STAGE 0
#endif

// load two adjacent bf16 as one aligned 32-bit word (callers guarantee
// even element index => 4-byte alignment)
__device__ __forceinline__ v2bf ldpair(const bf16* p)
{
    unsigned u = *(const unsigned*)p;
    return __builtin_bit_cast(v2bf, u);
}

// =====================================================================
// Generic complex GEMM on WMMA bf16 (f32 accumulate).
//   C[m,n] (+)= alpha * sum_k A[m,k]*B[k,n]   (complex)
// A planar bf16 row-major [M,K] (ldA); B planar bf16 stored TRANSPOSED,
// i.e. element (k,n) at Bt[n*ldB + k] -- K is contiguous for both
// operands, so all fragment loads merge into global_load_b128.
// One wave (32 threads) computes a 32x32 complex macro-tile = 2x2 WMMA
// tiles = 16 WMMAs per 32-wide k-step. Fragment lane layouts per CDNA5
// ISA 7.12.2 (16-bit A 16x32, B 32x16, f32 C/D 16x16).
// =====================================================================
template <int OUTBF16>
__global__ void __launch_bounds__(32)
cgemm_wmma_kernel(const bf16* __restrict__ Ar, const bf16* __restrict__ Ai,
                  long long aOff, int ldA, long long aInner, long long aOuter,
                  const bf16* __restrict__ Br, const bf16* __restrict__ Bi,
                  int ldB, long long bInner, long long bOuter,
                  float2* __restrict__ C, long long cOff, int ldCr, int ldCc,
                  long long cInner, long long cOuter,
                  bf16* __restrict__ Cr16, bf16* __restrict__ Ci16,
                  int innerCount, int Kc, float alpha, int beta)
{
    const int lane = threadIdx.x & 31;
    const int half = lane >> 4;
    const int r    = lane & 15;
    const int tileM = blockIdx.x * 32;
    const int tileN = blockIdx.y * 32;
    const int z  = blockIdx.z;
    const int zo = z / innerCount;
    const int zi = z % innerCount;
    const long long aBase = aOff + (long long)zo * aOuter + (long long)zi * aInner;
    const long long bBase =        (long long)zo * bOuter + (long long)zi * bInner;
    const long long cBase = cOff + (long long)zo * cOuter + (long long)zi * cInner;

    const v8f z8 = {0.f,0.f,0.f,0.f,0.f,0.f,0.f,0.f};
    v8f accRR[2][2] = {{z8, z8}, {z8, z8}};
    v8f accII[2][2] = {{z8, z8}, {z8, z8}};
    v8f accRI[2][2] = {{z8, z8}, {z8, z8}};

    for (int k0 = 0; k0 < Kc; k0 += 32) {
        v16bf arF[2], aiF[2], brF[2], biF[2];
#pragma unroll
        for (int t = 0; t < 2; ++t) {
#pragma unroll
            for (int v = 0; v < 8; ++v) {
                // A 16x32 bf16: lanes 0-15 -> K pairs {0..7 | 16..23},
                // lanes 16-31 -> {8..15 | 24..31}; row M = lane&15.
                const int ka = ((v < 4) ? (2 * v) : (16 + 2 * (v - 4))) + 8 * half;
                const long long aIdx =
                    aBase + (long long)(tileM + 16 * t + r) * ldA + (k0 + ka);
                const v2bf pr = ldpair(Ar + aIdx);
                const v2bf pi = ldpair(Ai + aIdx);
                arF[t][2*v] = pr[0];  arF[t][2*v+1] = pr[1];
                aiF[t][2*v] = pi[0];  aiF[t][2*v+1] = pi[1];

                // B 32x16 bf16: lanes 0-15 -> K pairs {0..7 | 16..23},
                // lanes 16-31 -> {8..15 | 24..31}; column N = lane&15.
                // B stored transposed -> the K pair is contiguous.
                const int kb = ((v < 4) ? (2 * v) : (16 + 2 * (v - 4))) + 8 * half;
                const long long b0 =
                    bBase + (long long)(tileN + 16 * t + r) * ldB + (k0 + kb);
                const v2bf qr = ldpair(Br + b0);
                const v2bf qi = ldpair(Bi + b0);
                brF[t][2*v] = qr[0];  brF[t][2*v+1] = qr[1];
                biF[t][2*v] = qi[0];  biF[t][2*v+1] = qi[1];
            }
        }

        // prefetch the next k-slab into cache (global_prefetch_b8)
        if (k0 + 32 < Kc) {
            const long long pA = aBase + (long long)(tileM + r) * ldA + (k0 + 32);
            const long long pB = bBase + (long long)(tileN + r) * ldB + (k0 + 32);
            __builtin_prefetch(Ar + pA, 0, 0);
            __builtin_prefetch(Ai + pA, 0, 0);
            __builtin_prefetch(Br + pB, 0, 0);
            __builtin_prefetch(Bi + pB, 0, 0);
        }

        // complex product via 4 real WMMAs per tile, f32 accumulate
#pragma unroll
        for (int mt = 0; mt < 2; ++mt) {
#pragma unroll
            for (int nt = 0; nt < 2; ++nt) {
                accRR[mt][nt] = __builtin_amdgcn_wmma_f32_16x16x32_bf16(
                    false, arF[mt], false, brF[nt], (short)0, accRR[mt][nt], false, false);
                accII[mt][nt] = __builtin_amdgcn_wmma_f32_16x16x32_bf16(
                    false, aiF[mt], false, biF[nt], (short)0, accII[mt][nt], false, false);
                accRI[mt][nt] = __builtin_amdgcn_wmma_f32_16x16x32_bf16(
                    false, arF[mt], false, biF[nt], (short)0, accRI[mt][nt], false, false);
                accRI[mt][nt] = __builtin_amdgcn_wmma_f32_16x16x32_bf16(
                    false, aiF[mt], false, brF[nt], (short)0, accRI[mt][nt], false, false);
            }
        }
    }

#pragma unroll
    for (int mt = 0; mt < 2; ++mt) {
#pragma unroll
        for (int nt = 0; nt < 2; ++nt) {
#pragma unroll
            for (int v = 0; v < 8; ++v) {
                // f32 C/D 16x16: VGPR v -> row M = v + 8*(lane>>4), col N = lane&15
                const int m = tileM + 16 * mt + v + 8 * half;
                const int n = tileN + 16 * nt + r;
                float re = (accRR[mt][nt][v] - accII[mt][nt][v]) * alpha;
                float im =  accRI[mt][nt][v]                     * alpha;
                const long long ci = cBase + (long long)m * ldCr + (long long)n * ldCc;
                if (OUTBF16) {
                    Cr16[ci] = (bf16)re;
                    Ci16[ci] = (bf16)im;
                } else {
                    if (beta) { float2 o = C[ci]; re += o.x; im += o.y; }
                    C[ci] = make_float2(re, im);
                }
            }
        }
    }
}

// ---------------- head conv: A = conv1d(r2c(X), hw) + hb ----------------
// X: [B,CIN,L,2] f32, W: [D,CIN,3,2], Bias: [D,2], out: [B,D,L] float2
__global__ void head_conv_kernel(const float* __restrict__ X, const float* __restrict__ W,
                                 const float* __restrict__ Bs, float2* __restrict__ out)
{
    int idx = blockIdx.x * blockDim.x + threadIdx.x;
    if (idx >= nB * nD * nL) return;
    int l = idx % nL;
    int o = (idx / nL) % nD;
    int b = idx / (nL * nD);
    float sr = Bs[2 * o], si = Bs[2 * o + 1];
    for (int i = 0; i < nK; ++i) {
        int lx = l + i - nK / 2;
        if (lx < 0 || lx >= nL) continue;
        for (int c = 0; c < nCIN; ++c) {
            const float* xp = X + ((long long)(b * nCIN + c) * nL + lx) * 2;
            const float* wp = W + ((long long)(o * nCIN + c) * nK + i) * 2;
            sr += xp[0] * wp[0] - xp[1] * wp[1];
            si += xp[0] * wp[1] + xp[1] * wp[0];
        }
    }
    out[idx] = make_float2(sr, si);
}

// ---------------- naive DFT along last dim (rows contiguous) ----------------
// Row is staged into LDS by the Tensor Data Mover (1-D tile DMA) when the
// toolchain exposes the TDM builtin; TENSORcnt + barrier order the handoff.
__global__ void dft_kernel(const float2* __restrict__ in, float2* __restrict__ out,
                           int N, int inverse)
{
    __shared__ float2 row[1024];
    const long long base = (long long)blockIdx.x * N;
#if USE_TDM_STAGE
    if (threadIdx.x < 32) {   // wave 0 issues the DMA (TDM ignores EXEC)
        const unsigned long long ga = (unsigned long long)(const void*)(in + base);
        // D# group0: count=1 | lds_addr (row[] is the only LDS object -> 0)
        //            | 57-bit global_addr | type=2 ("image")
        tdm_u4 g0;
        g0[0] = 1u;
        g0[1] = 0u;
        g0[2] = (unsigned)ga;
        g0[3] = ((unsigned)(ga >> 32) & 0x01ffffffu) | (2u << 30);
        // D# group1: wg_mask=0, data_size=3 (8B), no barrier/iterate/pad;
        // tensor_dim0 = N, tensor_dim1 = 1, tile_dim0 = N, tile_dim1 = 1,
        // tensor_dim0_stride = N.
        tdm_i8 g1;
        g1[0] = (int)(3u << 16);
        g1[1] = (int)((unsigned)N << 16);
        g1[2] = (int)(1u << 16);
        g1[3] = (int)((unsigned)N << 16);
        g1[4] = 1;
        g1[5] = N;
        g1[6] = 0;
        g1[7] = 0;
        tdm_i4 gz;
        gz[0] = 0; gz[1] = 0; gz[2] = 0; gz[3] = 0;   // <=2-D tensor
        tdm_i8 gz8;
        gz8[0] = 0; gz8[1] = 0; gz8[2] = 0; gz8[3] = 0;
        gz8[4] = 0; gz8[5] = 0; gz8[6] = 0; gz8[7] = 0;
        // 6-arg form (amdgpu-toolchain clang): (g0, g1, g2, g3, g4, cpol)
        __builtin_amdgcn_tensor_load_to_lds(g0, g1, gz, gz, gz8, 0);
        __builtin_amdgcn_s_wait_tensorcnt(0);
    }
    __syncthreads();
#else
    for (int i = threadIdx.x; i < N; i += blockDim.x) row[i] = in[base + i];
    __syncthreads();
#endif
    const float sgn   = inverse ? 1.0f : -1.0f;
    const float scale = inverse ? (1.0f / (float)N) : 1.0f;
    const float w0    = sgn * 2.0f * 3.14159265358979323846f / (float)N;
    for (int k = threadIdx.x; k < N; k += blockDim.x) {
        float ar = 0.f, ai = 0.f;
        for (int n = 0; n < N; ++n) {
            int kn = (int)(((long long)k * n) % N);
            float s, c;
            __sincosf(w0 * (float)kn, &s, &c);
            float2 x = row[n];
            ar += x.x * c - x.y * s;
            ai += x.x * s + x.y * c;
        }
        out[base + k] = make_float2(ar * scale, ai * scale);
    }
}

// ---------------- [B,D,L] f32c -> planar bf16 [B,L,D] ----------------
__global__ void transpose_cvt_kernel(const float2* __restrict__ in,
                                     bf16* __restrict__ outr, bf16* __restrict__ outi)
{
    int idx = blockIdx.x * blockDim.x + threadIdx.x;
    if (idx >= nB * nD * nL) return;
    int l = idx % nL;
    int d = (idx / nL) % nD;
    int b = idx / (nL * nD);
    float2 v = in[idx];
    long long o = ((long long)b * nL + l) * nD + d;
    outr[o] = (bf16)v.x;
    outi[o] = (bf16)v.y;
}

// -------- weights: f32 complex [S,R,C,2] -> TRANSPOSED planar bf16 [S][C][R] --------
__global__ void repack_wT_kernel(const float* __restrict__ in,
                                 bf16* __restrict__ outr, bf16* __restrict__ outi,
                                 int S, int R, int Cc)
{
    long long idx = (long long)blockIdx.x * blockDim.x + threadIdx.x;
    const long long total = (long long)S * R * Cc;
    if (idx >= total) return;
    int c  = (int)(idx % Cc);
    int r0 = (int)((idx / Cc) % R);
    int s  = (int)(idx / ((long long)Cc * R));
    float re = in[2 * idx];
    float im = in[2 * idx + 1];
    long long o = ((long long)s * Cc + c) * R + r0;
    outr[o] = (bf16)re;
    outi[o] = (bf16)im;
}

// -- ssca conv weights: in [D(o),D(c),3,2] -> planar [3][o][c] (transposed-B) --
__global__ void repack_conv_kernel(const float* __restrict__ W,
                                   bf16* __restrict__ outr, bf16* __restrict__ outi)
{
    int idx = blockIdx.x * blockDim.x + threadIdx.x;
    if (idx >= nK * nD * nD) return;
    int o = idx % nD;
    int c = (idx / nD) % nD;
    int i = idx / (nD * nD);
    const float* wp = W + ((long long)(o * nD + c) * nK + i) * 2;
    long long oi = ((long long)i * nD + o) * nD + c;   // [i][o][c]
    outr[oi] = (bf16)wp[0];
    outi[oi] = (bf16)wp[1];
}

// ---------------- magminmax over rows of length nL ----------------
__global__ void magminmax_kernel(const float2* __restrict__ S,
                                 bf16* __restrict__ outr, bf16* __restrict__ outi)
{
    __shared__ float smn[nL], smx[nL];
    const long long base = (long long)blockIdx.x * nL;
    int t = threadIdx.x;
    float2 v = S[base + t];
    float mag = sqrtf(v.x * v.x + v.y * v.y);
    smn[t] = mag; smx[t] = mag;
    __syncthreads();
    for (int off = nL / 2; off > 0; off >>= 1) {
        if (t < off) {
            smn[t] = fminf(smn[t], smn[t + off]);
            smx[t] = fmaxf(smx[t], smx[t + off]);
        }
        __syncthreads();
    }
    float mn = smn[0], mx = smx[0];
    float f = ((mag - mn) / (mx - mn + 1e-9f)) / (mag + 1e-9f);
    outr[base + t] = (bf16)(v.x * f);
    outi[base + t] = (bf16)(v.y * f);
}

// ---------------- ECA (pos-major Xc [B,L4,D]): column means ----------------
__global__ void col_mean_kernel(const float2* __restrict__ X, float2* __restrict__ mean)
{
    int idx = blockIdx.x * blockDim.x + threadIdx.x;
    if (idx >= nB * nD) return;
    int d = idx % nD;
    int b = idx / nD;
    const float2* src = X + (long long)b * nL4 * nD + d;
    float sr = 0.f, si = 0.f;
    for (int l = 0; l < nL4; ++l) {
        float2 v = src[(long long)l * nD];
        sr += v.x; si += v.y;
    }
    mean[idx] = make_float2(sr / nL4, si / nL4);
}

// -------- ECA gate+scale, write zero-padded planar bf16 [B, L4+2, D] --------
__global__ void eca_scale_pad_kernel(const float2* __restrict__ X,
                                     const float2* __restrict__ mean,
                                     const float* __restrict__ w,
                                     bf16* __restrict__ outr, bf16* __restrict__ outi)
{
    const int LP = nL4 + 2;
    long long idx = (long long)blockIdx.x * blockDim.x + threadIdx.x;
    const long long total = (long long)nB * LP * nD;
    if (idx >= total) return;
    int d  = (int)(idx % nD);
    int lp = (int)((idx / nD) % LP);
    int b  = (int)(idx / ((long long)nD * LP));
    if (lp == 0 || lp == LP - 1) { outr[idx] = (bf16)0.f; outi[idx] = (bf16)0.f; return; }
    int l = lp - 1;
    float zr = 0.f, zi = 0.f;
#pragma unroll
    for (int j = 0; j < nKE; ++j) {
        int cc = d + j - nKE / 2;
        if (cc < 0 || cc >= nD) continue;
        float2 m = mean[b * nD + cc];
        float wr = w[2 * j], wi = w[2 * j + 1];
        zr += m.x * wr - m.y * wi;
        zi += m.x * wi + m.y * wr;
    }
    float gr = 1.f / (1.f + __expf(-zr));
    float gi = 1.f / (1.f + __expf(-zi));
    float2 x = X[((long long)b * nL4 + l) * nD + d];
    outr[idx] = (bf16)(x.x * gr - x.y * gi);
    outi[idx] = (bf16)(x.x * gi + x.y * gr);
}

// ---------------- final ECA on concat([P;Q], channel axis) ----------------
__global__ void row_mean2_kernel(const float2* __restrict__ P, const float2* __restrict__ Q,
                                 float2* __restrict__ mean)
{
    int idx = blockIdx.x * blockDim.x + threadIdx.x;
    if (idx >= nB * 2 * nD) return;
    int ch = idx % (2 * nD);
    int b  = idx / (2 * nD);
    const float2* src = (ch < nD) ? (P + (long long)(b * nD + ch) * nL4)
                                  : (Q + (long long)(b * nD + ch - nD) * nL4);
    float sr = 0.f, si = 0.f;
    for (int l = 0; l < nL4; ++l) { sr += src[l].x; si += src[l].y; }
    mean[idx] = make_float2(sr / nL4, si / nL4);
}

__global__ void eca_final_kernel(const float2* __restrict__ P, const float2* __restrict__ Q,
                                 const float2* __restrict__ mean, const float* __restrict__ w,
                                 float2* __restrict__ dst)
{
    const int C2 = 2 * nD;
    long long idx = (long long)blockIdx.x * blockDim.x + threadIdx.x;
    const long long total = (long long)nB * C2 * nL4;
    if (idx >= total) return;
    int l  = (int)(idx % nL4);
    int ch = (int)((idx / nL4) % C2);
    int b  = (int)(idx / ((long long)nL4 * C2));
    float zr = 0.f, zi = 0.f;
#pragma unroll
    for (int j = 0; j < nKE; ++j) {
        int cc = ch + j - nKE / 2;
        if (cc < 0 || cc >= C2) continue;
        float2 m = mean[b * C2 + cc];
        float wr = w[2 * j], wi = w[2 * j + 1];
        zr += m.x * wr - m.y * wi;
        zi += m.x * wi + m.y * wr;
    }
    float gr = 1.f / (1.f + __expf(-zr));
    float gi = 1.f / (1.f + __expf(-zi));
    float2 x = (ch < nD) ? P[(long long)(b * nD + ch) * nL4 + l]
                         : Q[(long long)(b * nD + ch - nD) * nL4 + l];
    dst[idx] = make_float2(x.x * gr - x.y * gi, x.x * gi + x.y * gr);
}

// =====================================================================
// Host orchestration
// =====================================================================
extern "C" void kernel_launch(void* const* d_in, const int* in_sizes, int n_in,
                              void* d_out, int out_size, void* d_ws, size_t ws_size,
                              hipStream_t stream)
{
    (void)in_sizes; (void)n_in; (void)out_size; (void)ws_size;

    const float* X1       = (const float*)d_in[0];
    const float* X2       = (const float*)d_in[1];
    const float* head_w   = (const float*)d_in[2];   // [2,D,CIN,3,2]
    const float* head_b   = (const float*)d_in[3];   // [2,D,2]
    const float* wq_f     = (const float*)d_in[4];   // [2,4,D,HD,2]
    const float* wk_f     = (const float*)d_in[5];
    const float* wv_f     = (const float*)d_in[6];
    const float* wo_f     = (const float*)d_in[7];   // [2,4,HD,D,2]
    const float* ssca_eca = (const float*)d_in[8];   // [2,KE,2]
    const float* conv_w   = (const float*)d_in[9];   // [2,D,D,3,2]
    const float* top_eca  = (const float*)d_in[10];  // [2,KE,2]

    // ---- workspace bump allocator ----
    size_t off = 0;
    auto alloc = [&](size_t bytes) -> void* {
        off = (off + 255) & ~(size_t)255;
        void* p = (char*)d_ws + off;
        off += bytes;
        return p;
    };

    float2* cA   = (float2*)alloc((size_t)nB * nD * nL * sizeof(float2));
    float2* cBc  = (float2*)alloc((size_t)nB * nD * nL * sizeof(float2));
    float2* cAf  = (float2*)alloc((size_t)nB * nD * nL * sizeof(float2));
    float2* cBf  = (float2*)alloc((size_t)nB * nD * nL * sizeof(float2));

    // transposed weight planes: wqT/wkT/wvT: [8][HD][D]; woT: [8][D][HD]
    bf16* wqT_r = (bf16*)alloc((size_t)8 * nHD * nD * sizeof(bf16));
    bf16* wqT_i = (bf16*)alloc((size_t)8 * nHD * nD * sizeof(bf16));
    bf16* wkT_r = (bf16*)alloc((size_t)8 * nHD * nD * sizeof(bf16));
    bf16* wkT_i = (bf16*)alloc((size_t)8 * nHD * nD * sizeof(bf16));
    bf16* wvT_r = (bf16*)alloc((size_t)8 * nHD * nD * sizeof(bf16));
    bf16* wvT_i = (bf16*)alloc((size_t)8 * nHD * nD * sizeof(bf16));
    bf16* woT_r = (bf16*)alloc((size_t)8 * nD * nHD * sizeof(bf16));
    bf16* woT_i = (bf16*)alloc((size_t)8 * nD * nHD * sizeof(bf16));
    bf16* wtT_r = (bf16*)alloc((size_t)2 * nK * nD * nD * sizeof(bf16));
    bf16* wtT_i = (bf16*)alloc((size_t)2 * nK * nD * nD * sizeof(bf16));

    bf16* xa_r = (bf16*)alloc((size_t)nB * nL * nD * sizeof(bf16));
    bf16* xa_i = (bf16*)alloc((size_t)nB * nL * nD * sizeof(bf16));
    bf16* xb_r = (bf16*)alloc((size_t)nB * nL * nD * sizeof(bf16));
    bf16* xb_i = (bf16*)alloc((size_t)nB * nL * nD * sizeof(bf16));

    bf16* qb_r  = (bf16*)alloc((size_t)nB * nL * nHD * sizeof(bf16));
    bf16* qb_i  = (bf16*)alloc((size_t)nB * nL * nHD * sizeof(bf16));
    bf16* kb_r  = (bf16*)alloc((size_t)nB * nL * nHD * sizeof(bf16));
    bf16* kb_i  = (bf16*)alloc((size_t)nB * nL * nHD * sizeof(bf16));
    bf16* vbT_r = (bf16*)alloc((size_t)nB * nHD * nL * sizeof(bf16));  // [B, HD, L]
    bf16* vbT_i = (bf16*)alloc((size_t)nB * nHD * nL * sizeof(bf16));
    bf16* ob_r  = (bf16*)alloc((size_t)nB * nL * nHD * sizeof(bf16));
    bf16* ob_i  = (bf16*)alloc((size_t)nB * nL * nHD * sizeof(bf16));

    float2* sbuf = (float2*)alloc((size_t)nB * nH * nL * nL * sizeof(float2));
    bf16*   st_r = (bf16*)alloc((size_t)nB * nH * nL * nL * sizeof(bf16));
    bf16*   st_i = (bf16*)alloc((size_t)nB * nH * nL * nL * sizeof(bf16));

    float2* Xc     = (float2*)alloc((size_t)nB * nL4 * nD * sizeof(float2));
    float2* meanBD = (float2*)alloc((size_t)nB * nD * sizeof(float2));
    bf16*   xg_r   = (bf16*)alloc((size_t)nB * (nL4 + 2) * nD * sizeof(bf16));
    bf16*   xg_i   = (bf16*)alloc((size_t)nB * (nL4 + 2) * nD * sizeof(bf16));

    float2* Rb   = (float2*)alloc((size_t)nB * nD * nL4 * sizeof(float2));
    float2* KKb  = (float2*)alloc((size_t)nB * nD * nL4 * sizeof(float2));
    float2* FKK  = (float2*)alloc((size_t)nB * nD * nL4 * sizeof(float2));
    float2* IR   = (float2*)alloc((size_t)nB * nD * nL4 * sizeof(float2));
    float2* meanTop = (float2*)alloc((size_t)nB * 2 * nD * sizeof(float2));

    auto gemm = [&](const bf16* Ar_, const bf16* Ai_, long long aOff, int ldA,
                    long long aIn, long long aOut,
                    const bf16* Br_, const bf16* Bi_, int ldB,
                    long long bIn, long long bOut,
                    float2* C_, long long cOff, int ldCr, int ldCc,
                    long long cIn, long long cOut,
                    bf16* Cr_, bf16* Ci_, int outBf16,
                    int M, int N, int batch, int innerCount, int Kc,
                    float alpha, int beta) {
        dim3 g(M / 32, N / 32, batch), blk(32, 1, 1);
        if (outBf16) {
            cgemm_wmma_kernel<1><<<g, blk, 0, stream>>>(
                Ar_, Ai_, aOff, ldA, aIn, aOut, Br_, Bi_, ldB, bIn, bOut,
                C_, cOff, ldCr, ldCc, cIn, cOut, Cr_, Ci_, innerCount, Kc, alpha, beta);
        } else {
            cgemm_wmma_kernel<0><<<g, blk, 0, stream>>>(
                Ar_, Ai_, aOff, ldA, aIn, aOut, Br_, Bi_, ldB, bIn, bOut,
                C_, cOff, ldCr, ldCc, cIn, cOut, Cr_, Ci_, innerCount, Kc, alpha, beta);
        }
    };

    // ---- weight repacks (f32 complex -> transposed planar bf16) ----
    {
        long long n = (long long)8 * nD * nHD;
        int g = (int)((n + 255) / 256);
        repack_wT_kernel<<<g, 256, 0, stream>>>(wq_f, wqT_r, wqT_i, 8, nD, nHD);
        repack_wT_kernel<<<g, 256, 0, stream>>>(wk_f, wkT_r, wkT_i, 8, nD, nHD);
        repack_wT_kernel<<<g, 256, 0, stream>>>(wv_f, wvT_r, wvT_i, 8, nD, nHD);
        repack_wT_kernel<<<g, 256, 0, stream>>>(wo_f, woT_r, woT_i, 8, nHD, nD);
    }
    for (int idx = 0; idx < 2; ++idx) {
        int n = nK * nD * nD;
        repack_conv_kernel<<<(n + 255) / 256, 256, 0, stream>>>(
            conv_w + (size_t)idx * nD * nD * nK * 2,
            wtT_r + (size_t)idx * nK * nD * nD,
            wtT_i + (size_t)idx * nK * nD * nD);
    }

    // ---- head convolutions ----
    {
        int n = nB * nD * nL, g = (n + 255) / 256;
        head_conv_kernel<<<g, 256, 0, stream>>>(X1, head_w, head_b, cA);
        head_conv_kernel<<<g, 256, 0, stream>>>(X2, head_w + (size_t)nD * nCIN * nK * 2,
                                                head_b + 2 * nD, cBc);
    }

    // ---- ifft(A), ifft(Bc) for second SSCA branch ----
    dft_kernel<<<nB * nD, 256, 0, stream>>>(cA, cAf, nL, 1);
    dft_kernel<<<nB * nD, 256, 0, stream>>>(cBc, cBf, nL, 1);

    // ---- one SSCA pass ----
    auto run_ssca = [&](const float2* inA, const float2* inB, int idx, float2* outC) {
        int nT = nB * nD * nL;
        transpose_cvt_kernel<<<(nT + 255) / 256, 256, 0, stream>>>(inA, xa_r, xa_i);
        transpose_cvt_kernel<<<(nT + 255) / 256, 256, 0, stream>>>(inB, xb_r, xb_i);
        const float* ecaw = ssca_eca + idx * nKE * 2;

        for (int m = 0; m < 4; ++m) {
            const bf16 *qsr, *qsi, *ksr, *ksi;
            if      (m == 0) { qsr = xa_r; qsi = xa_i; ksr = xa_r; ksi = xa_i; }
            else if (m == 1) { qsr = xb_r; qsi = xb_i; ksr = xb_r; ksi = xb_i; }
            else if (m == 2) { qsr = xa_r; qsi = xa_i; ksr = xb_r; ksi = xb_i; }
            else             { qsr = xb_r; qsi = xb_i; ksr = xa_r; ksi = xa_i; }
            long long wofs = (long long)(idx * 4 + m) * nD * nHD;

            // q/k projections: [B*L, D] x W^T -> planar bf16 [B,L,HD]
            gemm(qsr, qsi, 0, nD, 0, 0, wqT_r + wofs, wqT_i + wofs, nD, 0, 0,
                 nullptr, 0, nHD, 1, 0, 0, qb_r, qb_i, 1,
                 nB * nL, nHD, 1, 1, nD, 1.0f, 0);
            gemm(ksr, ksi, 0, nD, 0, 0, wkT_r + wofs, wkT_i + wofs, nD, 0, 0,
                 nullptr, 0, nHD, 1, 0, 0, kb_r, kb_i, 1,
                 nB * nL, nHD, 1, 1, nD, 1.0f, 0);
            // v projection: store TRANSPOSED -> vT [B, HD, L]
            gemm(ksr, ksi, 0, nD, 0, (long long)nL * nD,
                 wvT_r + wofs, wvT_i + wofs, nD, 0, 0,
                 nullptr, 0, 1, nL, 0, (long long)nHD * nL,
                 vbT_r, vbT_i, 1,
                 nL, nHD, nB, 1, nD, 1.0f, 0);

            // scores: per (b,h)  q[L,DK] x k^T, scaled 1/sqrt(DK)
            gemm(qb_r, qb_i, 0, nHD, nDK, (long long)nL * nHD,
                 kb_r, kb_i, nHD, nDK, (long long)nL * nHD,
                 sbuf, 0, nL, 1, (long long)nL * nL, (long long)nH * nL * nL,
                 nullptr, nullptr, 0,
                 nL, nL, nB * nH, nH, nDK, 1.0f / sqrtf((float)nDK), 0);

            // magminmax over key dim
            magminmax_kernel<<<nB * nH * nL, nL, 0, stream>>>(sbuf, st_r, st_i);

            // attention out: per (b,h)  s~[L,L] x v (vT as transposed-B)
            gemm(st_r, st_i, 0, nL, (long long)nL * nL, (long long)nH * nL * nL,
                 vbT_r, vbT_i, nL, (long long)nDV * nL, (long long)nHD * nL,
                 nullptr, 0, nHD, 1, nDV, (long long)nL * nHD,
                 ob_r, ob_i, 1,
                 nL, nDV, nB * nH, nH, nL, 1.0f, 0);

            // Wo projection -> section m of Xc (pos-major [B, 4L, D])
            long long woofs = (long long)(idx * 4 + m) * nHD * nD;
            gemm(ob_r, ob_i, 0, nHD, 0, (long long)nL * nHD,
                 woT_r + woofs, woT_i + woofs, nHD, 0, 0,
                 Xc, (long long)m * nL * nD, nD, 1, 0, (long long)nL4 * nD,
                 nullptr, nullptr, 0,
                 nL, nD, nB, 1, nHD, 1.0f, 0);
        }

        // ECA over channels of Xc, write zero-padded planar bf16
        {
            int n = nB * nD;
            col_mean_kernel<<<(n + 255) / 256, 256, 0, stream>>>(Xc, meanBD);
        }
        {
            long long n = (long long)nB * (nL4 + 2) * nD;
            eca_scale_pad_kernel<<<(int)((n + 255) / 256), 256, 0, stream>>>(
                Xc, meanBD, ecaw, xg_r, xg_i);
        }

        // conv1d (K=3) as 3 shifted accumulated GEMMs, output channel-major
        for (int i = 0; i < nK; ++i) {
            gemm(xg_r, xg_i, (long long)i * nD, nD, 0, (long long)(nL4 + 2) * nD,
                 wtT_r + (long long)(idx * nK + i) * nD * nD,
                 wtT_i + (long long)(idx * nK + i) * nD * nD, nD, 0, 0,
                 outC, 0, 1, nL4, 0, (long long)nD * nL4,
                 nullptr, nullptr, 0,
                 nL4, nD, nB, 1, nD, 1.0f, (i > 0) ? 1 : 0);
        }
    };

    run_ssca(cA,  cBc, 0, Rb);    // R  = ssca(A, Bc)
    run_ssca(cAf, cBf, 1, KKb);   // kk = ssca(ifft A, ifft Bc)

    // fft(kk), ifft(R) along length-4L axis
    dft_kernel<<<nB * nD, 256, 0, stream>>>(KKb, FKK, nL4, 0);
    dft_kernel<<<nB * nD, 256, 0, stream>>>(Rb,  IR,  nL4, 1);

    // outputs: eca(concat([R, fft(kk)])), eca(concat([kk, ifft(R)]))
    float2* out0 = (float2*)d_out;
    float2* out1 = out0 + (long long)nB * 2 * nD * nL4;
    {
        int n = nB * 2 * nD;
        row_mean2_kernel<<<(n + 255) / 256, 256, 0, stream>>>(Rb, FKK, meanTop);
    }
    {
        long long n = (long long)nB * 2 * nD * nL4;
        eca_final_kernel<<<(int)((n + 255) / 256), 256, 0, stream>>>(
            Rb, FKK, meanTop, top_eca, out0);
    }
    {
        int n = nB * 2 * nD;
        row_mean2_kernel<<<(n + 255) / 256, 256, 0, stream>>>(KKb, IR, meanTop);
    }
    {
        long long n = (long long)nB * 2 * nD * nL4;
        eca_final_kernel<<<(int)((n + 255) / 256), 256, 0, stream>>>(
            KKb, IR, meanTop, top_eca + nKE * 2, out1);
    }
}
